// GNNEncoder_56272661512354
// MI455X (gfx1250) — compile-verified, matrix operand order checked by reading, not run on hardware
//
#include <hip/hip_runtime.h>

// ---------------------------------------------------------------------------
// Hetero-SAGE 2-layer GNN for MI455X (gfx1250, wave32, WMMA).
// Aggregation: fp32 global atomics (memory-bound, rides L2).
// GEMMs: v_wmma_f32_16x16x32_bf16, bf16 inputs / f32 accumulate.
//   - 2 M-tiles x 4 N-tiles per wave (8 WMMAs per A-fragment pair,
//     each LDS B fragment used twice)
//   - B panel staged block-cooperatively into LDS (ds_load_b128)
//   - A loads double-buffered via manual 2x k-unroll (no pipeline copies)
//   - __launch_bounds__(256,1): full VGPR budget, no spills
// ---------------------------------------------------------------------------

typedef __attribute__((ext_vector_type(16))) __bf16 v16bf;
typedef __attribute__((ext_vector_type(8)))  float  v8f;

#define N_USER 100000
#define N_BIZ  50000
#define NEDGE  500000
#define D_IN   256
#define D_H    512
#define D_OUT  256

#define LDS_B_BYTES 65536   // max ktiles(16) * 4 tiles * 1024 B/tile

// ---------------------------------------------------------------------------
// Edge scatter: for each (edge, 4-float chunk), gather src/dst features and
// atomically accumulate into the opposite side's segment-sum buffer.
// ---------------------------------------------------------------------------
__global__ void sage_scatter(const int* __restrict__ src, const int* __restrict__ dst,
                             const float* __restrict__ Xu,  // user-side features [*, D]
                             const float* __restrict__ Xb,  // biz-side features  [*, D]
                             float* __restrict__ AggB,      // sums into biz rows
                             float* __restrict__ AggU,      // sums into user rows
                             float* __restrict__ cntU, float* __restrict__ cntB,
                             int nEdges, int D, int chunkShift, int doDeg)
{
    long long idx = (long long)blockIdx.x * blockDim.x + threadIdx.x;
    long long total = (long long)nEdges << chunkShift;
    if (idx >= total) return;
    int c = (int)(idx & ((1 << chunkShift) - 1));
    int e = (int)(idx >> chunkShift);
    int s = src[e];
    int d = dst[e];
    if (doDeg && c == 0) {
        atomicAdd(cntB + d, 1.0f);
        atomicAdd(cntU + s, 1.0f);
    }
    const float4 xu = *(const float4*)(Xu + (size_t)s * D + (c << 2));
    const float4 xb = *(const float4*)(Xb + (size_t)d * D + (c << 2));
    float* ab = AggB + (size_t)d * D + (c << 2);
    float* au = AggU + (size_t)s * D + (c << 2);
    atomicAdd(ab + 0, xu.x); atomicAdd(ab + 1, xu.y);
    atomicAdd(ab + 2, xu.z); atomicAdd(ab + 3, xu.w);
    atomicAdd(au + 0, xb.x); atomicAdd(au + 1, xb.y);
    atomicAdd(au + 2, xb.z); atomicAdd(au + 3, xb.w);
}

// counts -> 1/max(count,1) in place
__global__ void inv_cnt(float* __restrict__ c, int n)
{
    int i = blockIdx.x * blockDim.x + threadIdx.x;
    if (i < n) c[i] = 1.0f / fmaxf(c[i], 1.0f);
}

// ---------------------------------------------------------------------------
// Pack a K x N fp32 weight matrix into bf16 laid out exactly as the CDNA5
// 16-bit B-fragment (ISA 7.12.2): per (ktile,ntile,lane) 16 contiguous bf16.
//   k = 32*t + (v>=4 ? 16 : 0) + (lane>=16 ? 8 : 0) + 2*(v&3) + p
//   n = 16*j + (lane & 15)
// ---------------------------------------------------------------------------
__global__ void pack_w_bf16(const float* __restrict__ W, __bf16* __restrict__ P,
                            int K, int N)
{
    int idx = blockIdx.x * blockDim.x + threadIdx.x;
    int ntiles = N >> 4;
    int ktiles = K >> 5;
    int total = ktiles * ntiles * 32;
    if (idx >= total) return;
    int lane = idx & 31;
    int rest = idx >> 5;
    int j = rest % ntiles;
    int t = rest / ntiles;
    int n  = (j << 4) + (lane & 15);
    int kb = (t << 5) + ((lane >= 16) ? 8 : 0);
    __bf16* out = P + (size_t)idx * 16;
#pragma unroll
    for (int e = 0; e < 16; ++e) {
        int v = e >> 1, p = e & 1;
        int k = kb + ((v >= 4) ? 16 : 0) + ((v & 3) << 1) + p;
        out[e] = (__bf16)W[(size_t)k * N + n];
    }
}

// ---------------------------------------------------------------------------
// Helpers for the GEMM kernel
// ---------------------------------------------------------------------------
__device__ __forceinline__ int koff8(int v)
{
    return ((v >= 4) ? 16 : 0) + ((v & 3) << 1);
}

// Stage the 4-tile-wide B panel (all ktiles) into LDS, 16B per copy.
// LDS layout: frag(t, i, lane) at byte offset t*4096 + (i*32 + lane)*32.
__device__ __forceinline__ void stage_B(const __bf16* __restrict__ P, char* lds,
                                        int ktiles, int ntiles, int tn0, int tid)
{
    const int units = ktiles << 8;          // #16-byte units (ktiles*4096/16)
    for (int u = tid; u < units; u += 256) {
        const int t   = u >> 8;
        const int rem = u & 255;
        const float4 v = *(const float4*)(P + (size_t)(t * ntiles + tn0) * 512 + rem * 8);
        *(float4*)(lds + t * 4096 + rem * 16) = v;
    }
}

// Convert one k-step's raw A floats to the bf16 A fragment (optionally scaled).
template <bool SCALE>
__device__ __forceinline__ void build_a(const float2* fa, float inv, v16bf& a)
{
#pragma unroll
    for (int v = 0; v < 8; ++v) {
        if (SCALE) {
            a[2 * v]     = (__bf16)(fa[v].x * inv);
            a[2 * v + 1] = (__bf16)(fa[v].y * inv);
        } else {
            a[2 * v]     = (__bf16)fa[v].x;
            a[2 * v + 1] = (__bf16)fa[v].y;
        }
    }
}

__device__ __forceinline__ void load_fa(const float* __restrict__ abase, int t,
                                        float2* fa)
{
    const float* an = abase + (t << 5);
#pragma unroll
    for (int v = 0; v < 8; ++v) fa[v] = *(const float2*)(an + koff8(v));
}

__device__ __forceinline__ void wmma8(const v16bf& a0, const v16bf& a1,
                                      const char* bf, v8f* acc0, v8f* acc1)
{
#pragma unroll
    for (int i = 0; i < 4; ++i) {
        const v16bf b = *(const v16bf*)(bf + i * 1024);   // load once, use twice
        acc0[i] = __builtin_amdgcn_wmma_f32_16x16x32_bf16(false, a0, false, b,
                                                          (short)0, acc0[i], false, false);
        acc1[i] = __builtin_amdgcn_wmma_f32_16x16x32_bf16(false, a1, false, b,
                                                          (short)0, acc1[i], false, false);
    }
}

// One K-sweep of the 2x4 register-blocked GEMM, k-loop manually unrolled 2x
// with explicit A double buffers (faA/faB) so the pipeline rotation is a
// compile-time rename instead of v_mov copies. ktiles must be even (8 or 16).
template <bool SCALE>
__device__ __forceinline__ void gemm_pass(const float* __restrict__ abase0,
                                          const float* __restrict__ abase1,
                                          float inv0, float inv1,
                                          const char* lds, int ktiles, int lane,
                                          v8f* acc0, v8f* acc1)
{
    float2 faA0[8], faA1[8], faB0[8], faB1[8];
    load_fa(abase0, 0, faA0);
    load_fa(abase1, 0, faA1);

    for (int t = 0; t < ktiles; t += 2) {
        // ---- even step: consume faA, prefetch faB for t+1 ----
        {
            v16bf a0, a1;
            build_a<SCALE>(faA0, inv0, a0);
            build_a<SCALE>(faA1, inv1, a1);
            load_fa(abase0, t + 1, faB0);          // t+1 < ktiles (ktiles even)
            load_fa(abase1, t + 1, faB1);
            wmma8(a0, a1, lds + t * 4096 + lane * 32, acc0, acc1);
        }
        // ---- odd step: consume faB, prefetch faA for t+2 ----
        {
            v16bf a0, a1;
            build_a<SCALE>(faB0, inv0, a0);
            build_a<SCALE>(faB1, inv1, a1);
            if (t + 2 < ktiles) {
                load_fa(abase0, t + 2, faA0);
                load_fa(abase1, t + 2, faA1);
            }
            wmma8(a0, a1, lds + (t + 1) * 4096 + lane * 32, acc0, acc1);
        }
    }
}

// ---------------------------------------------------------------------------
// Fused SAGE GEMM: Out = ReLU( (Agg .* invc) @ Wl + Xd @ Wr + bias )
// Block = 8 waves; all waves share one N-tile group (4 tiles); wave w owns
// M-tiles (mblk*16 + 2w) and (mblk*16 + 2w + 1).
// ---------------------------------------------------------------------------
__global__ void __launch_bounds__(256, 1)
sage_gemm24(const float* __restrict__ Agg, const float* __restrict__ invc,
            const float* __restrict__ Xd,
            const __bf16* __restrict__ Pl, const __bf16* __restrict__ Pr,
            const float* __restrict__ bias,
            float* __restrict__ Out, int M, int K, int N)
{
    __shared__ __align__(16) char ldsB[LDS_B_BYTES];

    const int tid  = threadIdx.x;
    const int lane = tid & 31;
    const int wave = tid >> 5;
    const int ngroups = N >> 6;                 // groups of 4 N-tiles
    const int mtiles  = M >> 4;
    const int tg   = blockIdx.x % ngroups;      // block-uniform N-tile group
    const int mblk = blockIdx.x / ngroups;
    const int tm0  = mblk * 16 + wave * 2;      // wave-uniform M-tiles
    const int tm1  = tm0 + 1;
    const bool act0 = (tm0 < mtiles);
    const bool act1 = (tm1 < mtiles);

    const int tn0    = tg << 2;
    const int ntiles = N >> 4;
    const int nLane  = lane & 15;
    const int rA0    = (tm0 << 4) + nLane;
    const int rA1    = rA0 + 16;
    const int kb     = (lane >= 16) ? 8 : 0;
    const int ktiles = K >> 5;

    v8f acc0[4], acc1[4];
#pragma unroll
    for (int i = 0; i < 4; ++i) {
        const float bv = bias[((tn0 + i) << 4) + nLane];
#pragma unroll
        for (int v = 0; v < 8; ++v) { acc0[i][v] = bv; acc1[i][v] = bv; }
    }

    // ======== pass 1: mean(Agg) @ Wl ========
    stage_B(Pl, ldsB, ktiles, ntiles, tn0, tid);
    __syncthreads();
    if (act0) {
        const float inv0 = invc[rA0];
        const float inv1 = act1 ? invc[rA1] : 1.0f;
        gemm_pass<true>(Agg + (size_t)rA0 * K + kb, Agg + (size_t)rA1 * K + kb,
                        inv0, inv1, ldsB, ktiles, lane, acc0, acc1);
    }
    __syncthreads();   // everyone done reading pass-1 B panel

    // ======== pass 2: Xd @ Wr ========
    stage_B(Pr, ldsB, ktiles, ntiles, tn0, tid);
    __syncthreads();
    if (act0) {
        gemm_pass<false>(Xd + (size_t)rA0 * K + kb, Xd + (size_t)rA1 * K + kb,
                         0.0f, 0.0f, ldsB, ktiles, lane, acc0, acc1);

        // ---- ReLU + store (C/D layout: VGPR v -> row m = v + (lane>=16 ? 8 : 0)) ----
        const int rowOff = (lane >= 16) ? 8 : 0;
#pragma unroll
        for (int i = 0; i < 4; ++i) {
            const int ncol = ((tn0 + i) << 4) + nLane;
#pragma unroll
            for (int v = 0; v < 8; ++v) {
                const float x0 = acc0[i][v];
                Out[(size_t)((tm0 << 4) + rowOff + v) * N + ncol] = x0 > 0.0f ? x0 : 0.0f;
            }
            if (act1) {
#pragma unroll
                for (int v = 0; v < 8; ++v) {
                    const float x1 = acc1[i][v];
                    Out[(size_t)((tm1 << 4) + rowOff + v) * N + ncol] = x1 > 0.0f ? x1 : 0.0f;
                }
            }
        }
    }
}

// ---------------------------------------------------------------------------
// Host-side orchestration (all on `stream`, graph-capture safe).
// ---------------------------------------------------------------------------
static inline size_t alignUp(size_t x) { return (x + 255) & ~(size_t)255; }

extern "C" void kernel_launch(void* const* d_in, const int* in_sizes, int n_in,
                              void* d_out, int out_size, void* d_ws, size_t ws_size,
                              hipStream_t stream)
{
    (void)in_sizes; (void)n_in; (void)out_size; (void)ws_size;

    const float* x_user  = (const float*)d_in[0];
    const float* x_biz   = (const float*)d_in[1];
    const int*   e_src   = (const int*)d_in[2];
    const int*   e_dst   = (const int*)d_in[3];
    const float* W1_l_ub = (const float*)d_in[4];
    const float* W1_r_ub = (const float*)d_in[5];
    const float* b1_ub   = (const float*)d_in[6];
    const float* W1_l_bu = (const float*)d_in[7];
    const float* W1_r_bu = (const float*)d_in[8];
    const float* b1_bu   = (const float*)d_in[9];
    const float* W2_l_ub = (const float*)d_in[10];
    const float* W2_r_ub = (const float*)d_in[11];
    const float* b2_ub   = (const float*)d_in[12];
    const float* W2_l_bu = (const float*)d_in[13];
    const float* W2_r_bu = (const float*)d_in[14];
    const float* b2_bu   = (const float*)d_in[15];

    float* o_user = (float*)d_out;                          // [N_USER, 256]
    float* o_biz  = (float*)d_out + (size_t)N_USER * D_OUT; // [N_BIZ, 256]

    // ---- workspace carve ----
    char* p = (char*)d_ws;
    size_t off = 0;
    float* cntU = (float*)(p + off); off += alignUp((size_t)N_USER * 4);
    float* cntB = (float*)(p + off); off += alignUp((size_t)N_BIZ * 4);
    float* aggU = (float*)(p + off); off += alignUp((size_t)N_USER * D_H * 4);
    float* aggB = (float*)(p + off); off += alignUp((size_t)N_BIZ * D_H * 4);
    float* hU   = (float*)(p + off); off += alignUp((size_t)N_USER * D_H * 4);
    float* hB   = (float*)(p + off); off += alignUp((size_t)N_BIZ * D_H * 4);
    const size_t wBytes = (size_t)256 * 512 * sizeof(__bf16); // same for all 8
    __bf16* pW[8];
    for (int i = 0; i < 8; ++i) { pW[i] = (__bf16*)(p + off); off += alignUp(wBytes); }
    // pW: 0=W1_l_ub 1=W1_r_ub 2=W1_l_bu 3=W1_r_bu 4=W2_l_ub 5=W2_r_ub 6=W2_l_bu 7=W2_r_bu

    // ---- pack weights into bf16 B-fragment layout ----
    {
        const int tot1 = (D_IN >> 5) * (D_H >> 4) * 32;   // 8192
        const int tot2 = (D_H >> 5) * (D_OUT >> 4) * 32;  // 8192
        const int blk = 256;
        pack_w_bf16<<<(tot1 + blk - 1) / blk, blk, 0, stream>>>(W1_l_ub, pW[0], D_IN, D_H);
        pack_w_bf16<<<(tot1 + blk - 1) / blk, blk, 0, stream>>>(W1_r_ub, pW[1], D_IN, D_H);
        pack_w_bf16<<<(tot1 + blk - 1) / blk, blk, 0, stream>>>(W1_l_bu, pW[2], D_IN, D_H);
        pack_w_bf16<<<(tot1 + blk - 1) / blk, blk, 0, stream>>>(W1_r_bu, pW[3], D_IN, D_H);
        pack_w_bf16<<<(tot2 + blk - 1) / blk, blk, 0, stream>>>(W2_l_ub, pW[4], D_H, D_OUT);
        pack_w_bf16<<<(tot2 + blk - 1) / blk, blk, 0, stream>>>(W2_r_ub, pW[5], D_H, D_OUT);
        pack_w_bf16<<<(tot2 + blk - 1) / blk, blk, 0, stream>>>(W2_l_bu, pW[6], D_H, D_OUT);
        pack_w_bf16<<<(tot2 + blk - 1) / blk, blk, 0, stream>>>(W2_r_bu, pW[7], D_H, D_OUT);
    }

    // ---- layer 1 aggregation ----
    hipMemsetAsync(cntU, 0, (size_t)N_USER * 4, stream);
    hipMemsetAsync(cntB, 0, (size_t)N_BIZ * 4, stream);
    hipMemsetAsync(aggU, 0, (size_t)N_USER * D_IN * 4, stream);
    hipMemsetAsync(aggB, 0, (size_t)N_BIZ * D_IN * 4, stream);
    {
        const long long total = (long long)NEDGE * (D_IN / 4);  // 32M threads
        const int blk = 256;
        const long long nblk = (total + blk - 1) / blk;
        sage_scatter<<<(unsigned)nblk, blk, 0, stream>>>(
            e_src, e_dst, x_user, x_biz, aggB, aggU, cntU, cntB,
            NEDGE, D_IN, 6 /*log2(256/4)*/, 1);
    }
    {
        const int blk = 256;
        inv_cnt<<<(N_USER + blk - 1) / blk, blk, 0, stream>>>(cntU, N_USER);
        inv_cnt<<<(N_BIZ + blk - 1) / blk, blk, 0, stream>>>(cntB, N_BIZ);
    }

    // ---- layer 1 GEMMs: h = ReLU(mean @ Wl + x @ Wr + b) ----
    {
        const int blk = 256;                 // 8 waves/block
        const int mbB = (N_BIZ / 16 + 15) / 16, ngH = D_H / 64;
        sage_gemm24<<<mbB * ngH, blk, 0, stream>>>(
            aggB, cntB, x_biz, pW[0], pW[1], b1_ub, hB, N_BIZ, D_IN, D_H);
        const int mbU = (N_USER / 16 + 15) / 16;
        sage_gemm24<<<mbU * ngH, blk, 0, stream>>>(
            aggU, cntU, x_user, pW[2], pW[3], b1_bu, hU, N_USER, D_IN, D_H);
    }

    // ---- layer 2 aggregation (D = 512) ----
    hipMemsetAsync(aggU, 0, (size_t)N_USER * D_H * 4, stream);
    hipMemsetAsync(aggB, 0, (size_t)N_BIZ * D_H * 4, stream);
    {
        const long long total = (long long)NEDGE * (D_H / 4);   // 64M threads
        const int blk = 256;
        const long long nblk = (total + blk - 1) / blk;
        sage_scatter<<<(unsigned)nblk, blk, 0, stream>>>(
            e_src, e_dst, hU, hB, aggB, aggU, cntU, cntB,
            NEDGE, D_H, 7 /*log2(512/4)*/, 0);
    }

    // ---- layer 2 GEMMs -> outputs ----
    {
        const int blk = 256;
        const int mbB = (N_BIZ / 16 + 15) / 16, ngO = D_OUT / 64;
        sage_gemm24<<<mbB * ngO, blk, 0, stream>>>(
            aggB, cntB, hB, pW[4], pW[5], b2_ub, o_biz, N_BIZ, D_H, D_OUT);
        const int mbU = (N_USER / 16 + 15) / 16;
        sage_gemm24<<<mbU * ngO, blk, 0, stream>>>(
            aggU, cntU, hU, pW[6], pW[7], b2_bu, o_user, N_USER, D_H, D_OUT);
    }
}
